// UniversalCalculator_40862318854750
// MI455X (gfx1250) — matrix operand order repaired and checked by reading, not compiled, not run on hardware
//
#include <hip/hip_runtime.h>

#define N_TOKENS  8192
#define D_MODEL   512
#define D_FF      1024
#define N_EXPERTS 8
#define SCORE_SCALE 1.0f

#define TM 64    // tokens per workgroup tile
#define FB 128   // d_ff chunk width

typedef __attribute__((ext_vector_type(16))) __bf16 v16bf;
typedef __attribute__((ext_vector_type(8)))  __bf16 v8bf;
typedef __attribute__((ext_vector_type(4)))  __bf16 v4bf;
typedef __attribute__((ext_vector_type(8)))  float  v8f;
typedef __attribute__((ext_vector_type(4)))  float  v4f;

union FragAB { v16bf v; v8bf h[2]; };

// A fragment (16x32 bf16, M x K), row-major source with row stride `stride`.
// Per ISA layout: lanes 0-15 -> M=0..15, VGPR0-3 = K 0..7, VGPR4-7 = K 16..23;
// lanes 16-31 same M rows, K ranges 8..15 and 24..31.
__device__ __forceinline__ v16bf load_frag_a(const __bf16* base, int stride, int lane) {
    FragAB f;
    int m  = lane & 15;
    int kh = lane >> 4;
    const __bf16* p = base + (size_t)m * stride + kh * 8;
    f.h[0] = *(const v8bf*)(p);
    f.h[1] = *(const v8bf*)(p + 16);
    return f.v;
}

// B fragment (32x16 bf16, K x N) from a [N][K] (K-contiguous) matrix.
// Per ISA B layout: lanes 0-15 (col = lane) hold K 0..7 and 16..23; lanes 16-31 hold K 8..15 and 24..31.
__device__ __forceinline__ v16bf load_frag_b(const __bf16* base, int stride, int lane) {
    FragAB f;
    int n  = lane & 15;
    int kh = lane >> 4;
    const __bf16* p = base + (size_t)n * stride + kh * 8;
    f.h[0] = *(const v8bf*)(p);
    f.h[1] = *(const v8bf*)(p + 16);
    return f.v;
}

// ---------------- routing: build per-expert token lists ----------------
__global__ void route_kernel(const int* __restrict__ tidx, const float* __restrict__ tsc,
                             int* __restrict__ counts, int* __restrict__ tids,
                             float* __restrict__ tw) {
    int t = blockIdx.x * blockDim.x + threadIdx.x;
    if (t >= N_TOKENS) return;
    int   e0 = tidx[2 * t],             e1 = tidx[2 * t + 1];
    float s0 = tsc[2 * t] * SCORE_SCALE, s1 = tsc[2 * t + 1] * SCORE_SCALE;
    if (e0 == e1) {
        int p = atomicAdd(&counts[e0], 1);
        tids[e0 * N_TOKENS + p] = t; tw[e0 * N_TOKENS + p] = s0 + s1;
    } else {
        int p0 = atomicAdd(&counts[e0], 1);
        tids[e0 * N_TOKENS + p0] = t; tw[e0 * N_TOKENS + p0] = s0;
        int p1 = atomicAdd(&counts[e1], 1);
        tids[e1 * N_TOKENS + p1] = t; tw[e1 * N_TOKENS + p1] = s1;
    }
}

// ---------------- x: f32 -> bf16 ----------------
__global__ void cvt_x_kernel(const float* __restrict__ x, __bf16* __restrict__ xb, int n) {
    int i = (blockIdx.x * blockDim.x + threadIdx.x) * 4;
    if (i >= n) return;
    v4f v = *(const v4f*)(x + i);
    v4bf o;
    o[0] = (__bf16)v[0]; o[1] = (__bf16)v[1]; o[2] = (__bf16)v[2]; o[3] = (__bf16)v[3];
    *(v4bf*)(xb + i) = o;
}

// ---------------- batched transpose + f32->bf16 : [R][C] -> [C][R] ----------------
__global__ void transpose_cvt_kernel(const float* __restrict__ in, __bf16* __restrict__ out,
                                     int R, int C) {
    __shared__ float tile[32][33];
    const float* src = in  + (size_t)blockIdx.z * R * C;
    __bf16*      dst = out + (size_t)blockIdx.z * R * C;
    int tx = threadIdx.x, ty = threadIdx.y;        // 32 x 8
    int r0 = blockIdx.y * 32, c0 = blockIdx.x * 32;
    #pragma unroll
    for (int k = 0; k < 32; k += 8)
        tile[ty + k][tx] = src[(size_t)(r0 + ty + k) * C + (c0 + tx)];
    __syncthreads();
    #pragma unroll
    for (int k = 0; k < 32; k += 8)
        dst[(size_t)(c0 + ty + k) * R + (r0 + tx)] = (__bf16)tile[tx][ty + k];
}

// ---------------- main routed SwiGLU expert kernel ----------------
__global__ __launch_bounds__(256) void moe_expert_kernel(
    const __bf16* __restrict__ Xb,    // [N][D]
    const __bf16* __restrict__ WgT,   // [E][F][D]  (K=D contiguous)
    const __bf16* __restrict__ WuT,   // [E][F][D]
    const __bf16* __restrict__ WdT,   // [E][D][F]  (K=F contiguous)
    const int*    __restrict__ counts,
    const int*    __restrict__ tids,
    const float*  __restrict__ tw,
    float*        __restrict__ y)     // [N][D], pre-zeroed; atomic combine
{
    int e   = blockIdx.y;
    int cnt = counts[e];
    int t0  = blockIdx.x * TM;
    if (t0 >= cnt) return;

    __shared__ __bf16 Xt[TM][D_MODEL];   // 64 KB gathered token tile
    __shared__ __bf16 Hs[TM][FB];        // 16 KB SwiGLU intermediate chunk
    __shared__ int    tok[TM];
    __shared__ float  wgt[TM];

    int tid = threadIdx.x;
    if (tid < TM) {
        int s = t0 + tid;
        if (s < cnt) { tok[tid] = tids[e * N_TOKENS + s]; wgt[tid] = tw[e * N_TOKENS + s]; }
        else         { tok[tid] = 0;                      wgt[tid] = 0.0f; }
    }
    __syncthreads();

    // gather 64 token rows into LDS (4 threads per row, 16B vectors)
    {
        int r = tid >> 2, q = tid & 3;
        const v8bf* src = (const v8bf*)(Xb + (size_t)tok[r] * D_MODEL + q * 128);
        v8bf*       dst = (v8bf*)(&Xt[r][q * 128]);
        #pragma unroll
        for (int i = 0; i < 16; ++i) dst[i] = src[i];
    }
    __syncthreads();

    int wave = tid >> 5, lane = tid & 31;
    int mt  = wave & 3;              // token row-tile (4 x 16 rows)
    int nb1 = (wave >> 2) * 4;       // GEMM1 col-tile base (8 tiles over FB)
    int nb2 = (wave >> 2) * 16;      // GEMM2 col-tile base (32 tiles over D)
    int kh  = lane >> 4;
    int nlo = lane & 15;

    const __bf16* Wg_e = WgT + (size_t)e * D_FF * D_MODEL;
    const __bf16* Wu_e = WuT + (size_t)e * D_FF * D_MODEL;
    const __bf16* Wd_e = WdT + (size_t)e * D_MODEL * D_FF;

    const v8f zero8 = {0.f, 0.f, 0.f, 0.f, 0.f, 0.f, 0.f, 0.f};
    v8f yacc[16];
    #pragma unroll
    for (int j = 0; j < 16; ++j) yacc[j] = zero8;

    for (int fb = 0; fb < D_FF; fb += FB) {
        // ---- GEMM1: gate & up, M=64 N=128 K=512 ----
        v8f accG[4], accU[4];
        #pragma unroll
        for (int j = 0; j < 4; ++j) { accG[j] = zero8; accU[j] = zero8; }

        #pragma unroll 4
        for (int kk = 0; kk < D_MODEL; kk += 32) {
            v16bf a = load_frag_a(&Xt[mt * 16][kk], D_MODEL, lane);
            #pragma unroll
            for (int j = 0; j < 4; ++j) {
                size_t ncol = (size_t)(fb + (nb1 + j) * 16);
                v16bf bg = load_frag_b(Wg_e + ncol * D_MODEL + kk, D_MODEL, lane);
                v16bf bu = load_frag_b(Wu_e + ncol * D_MODEL + kk, D_MODEL, lane);
                accG[j] = __builtin_amdgcn_wmma_f32_16x16x32_bf16(
                    false, a, false, bg, (short)0, accG[j], false, false);
                accU[j] = __builtin_amdgcn_wmma_f32_16x16x32_bf16(
                    false, a, false, bu, (short)0, accU[j], false, false);
            }
        }

        // ---- fused SiLU(gate) * up -> bf16 H tile in LDS ----
        #pragma unroll
        for (int j = 0; j < 4; ++j) {
            #pragma unroll
            for (int i = 0; i < 8; ++i) {
                float g = accG[j][i];
                float u = accU[j][i];
                float h = (g / (1.0f + __expf(-g))) * u;
                Hs[mt * 16 + i + 8 * kh][(nb1 + j) * 16 + nlo] = (__bf16)h;
            }
        }
        __syncthreads();

        // ---- GEMM2: yacc += H @ Wd chunk, M=64 N=512 K=128 ----
        #pragma unroll
        for (int kk = 0; kk < FB; kk += 32) {
            v16bf a2 = load_frag_a(&Hs[mt * 16][kk], FB, lane);
            #pragma unroll
            for (int j = 0; j < 16; ++j) {
                size_t ncol = (size_t)((nb2 + j) * 16);
                v16bf bd = load_frag_b(Wd_e + ncol * D_FF + fb + kk, D_FF, lane);
                yacc[j] = __builtin_amdgcn_wmma_f32_16x16x32_bf16(
                    false, a2, false, bd, (short)0, yacc[j], false, false);
            }
        }
        __syncthreads();   // H reused next chunk
    }

    // ---- combine: scale by routing weight, atomic scatter-add into y ----
    #pragma unroll
    for (int j = 0; j < 16; ++j) {
        int ncol = (nb2 + j) * 16 + nlo;
        #pragma unroll
        for (int i = 0; i < 8; ++i) {
            int r = mt * 16 + i + 8 * kh;
            float val = yacc[j][i] * wgt[r];
            atomicAdd(&y[(size_t)tok[r] * D_MODEL + ncol], val);
        }
    }
}

// ---------------- host launcher ----------------
extern "C" void kernel_launch(void* const* d_in, const int* in_sizes, int n_in,
                              void* d_out, int out_size, void* d_ws, size_t ws_size,
                              hipStream_t stream) {
    (void)in_sizes; (void)n_in; (void)ws_size;
    const float* x    = (const float*)d_in[0];
    const int*   tidx = (const int*)  d_in[1];
    const float* tsc  = (const float*)d_in[2];
    const float* Wg   = (const float*)d_in[3];
    const float* Wu   = (const float*)d_in[4];
    const float* Wd   = (const float*)d_in[5];
    float* y = (float*)d_out;

    char* ws = (char*)d_ws;
    size_t o = 0;
    int*    counts = (int*)   (ws + o); o += 256;
    int*    tids   = (int*)   (ws + o); o += (size_t)N_EXPERTS * N_TOKENS * sizeof(int);
    float*  tw     = (float*) (ws + o); o += (size_t)N_EXPERTS * N_TOKENS * sizeof(float);
    __bf16* Xb     = (__bf16*)(ws + o); o += (size_t)N_TOKENS * D_MODEL * 2;
    __bf16* WgT    = (__bf16*)(ws + o); o += (size_t)N_EXPERTS * D_MODEL * D_FF * 2;
    __bf16* WuT    = (__bf16*)(ws + o); o += (size_t)N_EXPERTS * D_MODEL * D_FF * 2;
    __bf16* WdT    = (__bf16*)(ws + o); o += (size_t)N_EXPERTS * D_MODEL * D_FF * 2;

    hipMemsetAsync(counts, 0, 256, stream);
    hipMemsetAsync(y, 0, (size_t)out_size * sizeof(float), stream);

    route_kernel<<<(N_TOKENS + 255) / 256, 256, 0, stream>>>(tidx, tsc, counts, tids, tw);
    cvt_x_kernel<<<(N_TOKENS * D_MODEL / 4 + 255) / 256, 256, 0, stream>>>(
        x, Xb, N_TOKENS * D_MODEL);

    dim3 tb(32, 8);
    // W_gate/W_up: [E][D][F] -> [E][F][D]
    transpose_cvt_kernel<<<dim3(D_FF / 32, D_MODEL / 32, N_EXPERTS), tb, 0, stream>>>(
        Wg, WgT, D_MODEL, D_FF);
    transpose_cvt_kernel<<<dim3(D_FF / 32, D_MODEL / 32, N_EXPERTS), tb, 0, stream>>>(
        Wu, WuT, D_MODEL, D_FF);
    // W_down: [E][F][D] -> [E][D][F]
    transpose_cvt_kernel<<<dim3(D_MODEL / 32, D_FF / 32, N_EXPERTS), tb, 0, stream>>>(
        Wd, WdT, D_FF, D_MODEL);

    dim3 grid(N_TOKENS / TM, N_EXPERTS);
    moe_expert_kernel<<<grid, 256, 0, stream>>>(Xb, WgT, WuT, WdT, counts, tids, tw, y);
}
